// TemporalSSM_9345848836690
// MI455X (gfx1250) — matrix operand order repaired and checked by reading, not compiled, not run on hardware
//
#include <hip/hip_runtime.h>
#include <hip/hip_bf16.h>

// ---------------- CDNA5 WMMA types ----------------
typedef _Float16 v16h __attribute__((ext_vector_type(16)));
typedef float    v8f  __attribute__((ext_vector_type(8)));

#define SIN_DIM 128
#define T_DIM   256
#define EMBED   1024
#define STATE   64
#define BATCH   8
#define SEQ     2048
#define NROWS   (BATCH*SEQ)      // 16384
#define CHUNK   64
#define NCHUNK  (SEQ/CHUNK)      // 32

__device__ __forceinline__ float softplusf(float x) {
    return (x > 20.0f) ? x : log1pf(expf(x));
}

// ---------------------------------------------------------------------------
// Kernel 1: hypernetwork (tiny; 1 block).  t = silu(te@W1+b1)@W2+b2,
// then delta, A_bar, pow-table A_bar^j, and D.
// ---------------------------------------------------------------------------
__global__ __launch_bounds__(256) void k_hyper(
    const float* __restrict__ te,
    const float* __restrict__ Wl1, const float* __restrict__ bl1,
    const float* __restrict__ Wl2, const float* __restrict__ bl2,
    const float* __restrict__ W_A, const float* __restrict__ b_A,
    const float* __restrict__ W_dl, const float* __restrict__ b_dl,
    const float* __restrict__ W_D, const float* __restrict__ b_D,
    float* __restrict__ tbuf, float* __restrict__ deltaBuf,
    float* __restrict__ abarBuf, float* __restrict__ powtab,
    float* __restrict__ Dbuf)
{
    __shared__ float s_te[SIN_DIM];
    __shared__ float s_h1[T_DIM];
    __shared__ float s_t[T_DIM];
    const int tid = threadIdx.x;
    if (tid < SIN_DIM) s_te[tid] = te[tid];
    __syncthreads();

    float acc = bl1[tid];
    #pragma unroll 8
    for (int k = 0; k < SIN_DIM; ++k) acc = fmaf(s_te[k], Wl1[k*T_DIM + tid], acc);
    acc = acc / (1.0f + expf(-acc));          // silu
    s_h1[tid] = acc;
    __syncthreads();

    float t = bl2[tid];
    #pragma unroll 8
    for (int k = 0; k < T_DIM; ++k) t = fmaf(s_h1[k], Wl2[k*T_DIM + tid], t);
    s_t[tid] = t;
    tbuf[tid] = t;
    __syncthreads();

    if (tid < STATE) {
        const int s = tid;
        float aA = b_A[s], ad = b_dl[s];
        #pragma unroll 8
        for (int k = 0; k < T_DIM; ++k) {
            aA = fmaf(s_t[k], W_A[k*STATE + s], aA);
            ad = fmaf(s_t[k], W_dl[k*STATE + s], ad);
        }
        const float Adiag = -softplusf(aA);
        const float dl    = softplusf(ad) + 1e-4f;
        const float Abar  = expf(dl * Adiag);
        deltaBuf[s] = dl;
        abarBuf[s]  = Abar;
        float p = 1.0f;
        powtab[s*(CHUNK+1) + 0] = 1.0f;
        for (int j = 1; j <= CHUNK; ++j) { p *= Abar; powtab[s*(CHUNK+1) + j] = p; }
    }
    for (int e = tid; e < EMBED; e += 256) {
        float d = b_D[e];
        #pragma unroll 8
        for (int k = 0; k < T_DIM; ++k) d = fmaf(s_t[k], W_D[k*EMBED + e], d);
        Dbuf[e] = d;
    }
}

// ---------------------------------------------------------------------------
// Kernel 2: contract hypernet over W_B / W_C (128 MB read -> HBM-bound),
// write results pre-swizzled into WMMA B-operand lane layout (f16).
//   GEMM1 B operand: K=e (1024), N=s (64)   -> Bsw[(kc*4+nt)*32+lane][i]
//   GEMM2 B operand: K=s (64),   N=e (1024) -> Csw[(kc*64+nt)*32+lane][i]
// B-layout rule (16-bit, wave32): b[i] = Btile[K=(i&7)+8*(lane/16)+16*(i>>3)][N=lane%16]
// ---------------------------------------------------------------------------
__global__ __launch_bounds__(256) void k_BC(
    const float* __restrict__ tbuf, const float* __restrict__ deltaBuf,
    const float* __restrict__ W_B, const float* __restrict__ b_B,
    const float* __restrict__ W_C, const float* __restrict__ b_C,
    _Float16* __restrict__ Bsw, _Float16* __restrict__ Csw)
{
    __shared__ float s_t[T_DIM];
    s_t[threadIdx.x] = tbuf[threadIdx.x];
    __syncthreads();

    const int blk = blockIdx.x;
    if (blk < 256) {                      // ---- C[e][s] = sum_k t[k]*W_C[k][e][s]
        const int o = blk*256 + threadIdx.x;      // o == e*64+s (coalesced over k)
        float acc = b_C[o];
        #pragma unroll 4
        for (int k = 0; k < T_DIM; ++k)
            acc = fmaf(s_t[k], W_C[(size_t)k*(EMBED*STATE) + o], acc);
        const int e = o >> 6, s = o & 63;
        const int kc = s >> 5, lane = (((s >> 3) & 1) << 4) | (e & 15);
        const int i  = (s & 7) | (((s >> 4) & 1) << 3), nt = e >> 4;
        Csw[(size_t)((kc*64 + nt)*32 + lane)*16 + i] = (_Float16)acc;
    } else {                              // ---- B_bar[s][e] = delta[s]*(t@W_B + b_B)
        const int o = (blk - 256)*256 + threadIdx.x;  // o == s*1024+e
        float acc = b_B[o];
        #pragma unroll 4
        for (int k = 0; k < T_DIM; ++k)
            acc = fmaf(s_t[k], W_B[(size_t)k*(STATE*EMBED) + o], acc);
        const int s = o >> 10, e = o & 1023;
        const float bb = deltaBuf[s] * acc;
        const int kc = e >> 5, lane = (((e >> 3) & 1) << 4) | (s & 15);
        const int i  = (e & 7) | (((e >> 4) & 1) << 3), nt = s >> 4;
        Bsw[(size_t)((kc*4 + nt)*32 + lane)*16 + i] = (_Float16)bb;
    }
}

// ---------------------------------------------------------------------------
// Kernel 3: GEMM1  U[16384,64] = X[16384,1024] @ B_bar^T  via WMMA f16.
// 8 waves/block, each wave: 16 rows x full N=64 (4 tiles), K-loop 32x32.
// A-layout rule: a[0..7] <-> K = k0+8*half+(0..7); a[8..15] <-> +16.
// ---------------------------------------------------------------------------
__global__ __launch_bounds__(256) void k_gemm1(
    const float* __restrict__ x, const _Float16* __restrict__ Bsw,
    float* __restrict__ U)
{
    const int wave = threadIdx.x >> 5, lane = threadIdx.x & 31;
    const int half = lane >> 4, m = lane & 15;
    const int r0 = (blockIdx.x*8 + wave)*16;
    const float* xrow = x + (size_t)(r0 + m)*EMBED;
    const v16h* bp = (const v16h*)Bsw;

    v8f acc0 = {}, acc1 = {}, acc2 = {}, acc3 = {};
    for (int kc = 0; kc < 32; ++kc) {
        const float4* p = (const float4*)(xrow + kc*32 + 8*half);
        const float4 fa = p[0], fb = p[1], fc = p[4], fd = p[5];
        v16h a;
        a[0]=(_Float16)fa.x;  a[1]=(_Float16)fa.y;  a[2]=(_Float16)fa.z;  a[3]=(_Float16)fa.w;
        a[4]=(_Float16)fb.x;  a[5]=(_Float16)fb.y;  a[6]=(_Float16)fb.z;  a[7]=(_Float16)fb.w;
        a[8]=(_Float16)fc.x;  a[9]=(_Float16)fc.y;  a[10]=(_Float16)fc.z; a[11]=(_Float16)fc.w;
        a[12]=(_Float16)fd.x; a[13]=(_Float16)fd.y; a[14]=(_Float16)fd.z; a[15]=(_Float16)fd.w;

        const v16h b0 = bp[(kc*4 + 0)*32 + lane];
        const v16h b1 = bp[(kc*4 + 1)*32 + lane];
        const v16h b2 = bp[(kc*4 + 2)*32 + lane];
        const v16h b3 = bp[(kc*4 + 3)*32 + lane];
        acc0 = __builtin_amdgcn_wmma_f32_16x16x32_f16(false, a, false, b0, (short)0, acc0, false, false);
        acc1 = __builtin_amdgcn_wmma_f32_16x16x32_f16(false, a, false, b1, (short)0, acc1, false, false);
        acc2 = __builtin_amdgcn_wmma_f32_16x16x32_f16(false, a, false, b2, (short)0, acc2, false, false);
        acc3 = __builtin_amdgcn_wmma_f32_16x16x32_f16(false, a, false, b3, (short)0, acc3, false, false);
    }
    #pragma unroll
    for (int v = 0; v < 8; ++v) {          // D-layout: M = v + 8*half, N = m
        float* up = U + (size_t)(r0 + v + 8*half)*STATE + m;
        up[0]  = acc0[v]; up[16] = acc1[v]; up[32] = acc2[v]; up[48] = acc3[v];
    }
}

// ---------------------------------------------------------------------------
// Kernel 4: chunk-local scans (16384 parallel (b,s,chunk) threads).
// ---------------------------------------------------------------------------
__global__ __launch_bounds__(256) void k_scan1(
    const float* __restrict__ U, const float* __restrict__ abar,
    float* __restrict__ Hloc, float* __restrict__ CE)
{
    const int g = blockIdx.x*256 + threadIdx.x;   // 16384
    const int s = g & 63, bc = g >> 6;            // bc = b*32+c? -> b fast here
    const int b = bc & 7, c = bc >> 3;
    const float A = abar[s];
    float h = 0.0f;
    const size_t base = ((size_t)(b*SEQ + c*CHUNK))*STATE + s;
    #pragma unroll 4
    for (int i = 0; i < CHUNK; ++i) {
        h = fmaf(A, h, U[base + (size_t)i*STATE]);
        Hloc[base + (size_t)i*STATE] = h;
    }
    CE[(b*NCHUNK + c)*STATE + s] = h;
}

// Kernel 5: carry scan across chunks (512 channels x 32 steps).
__global__ __launch_bounds__(512) void k_scan2(
    const float* __restrict__ CE, const float* __restrict__ powtab,
    float* __restrict__ carryIn)
{
    const int tid = threadIdx.x, s = tid & 63, b = tid >> 6;
    const float A64 = powtab[s*(CHUNK+1) + CHUNK];
    float carry = 0.0f;
    for (int c = 0; c < NCHUNK; ++c) {
        const int idx = (b*NCHUNK + c)*STATE + s;
        carryIn[idx] = carry;
        carry = fmaf(A64, carry, CE[idx]);
    }
}

// ---------------------------------------------------------------------------
// Kernel 6: fix-up (h += A^(i+1)*carry) fused with f16 conversion and
// pre-swizzle into the GEMM2 WMMA A-operand lane layout (64B/lane store).
// ---------------------------------------------------------------------------
__global__ __launch_bounds__(256) void k_fixswz(
    const float* __restrict__ Hloc, const float* __restrict__ powtab,
    const float* __restrict__ carryIn, _Float16* __restrict__ Hsw)
{
    const int wave = threadIdx.x >> 5, lane = threadIdx.x & 31;
    const int half = lane >> 4, m = lane & 15;
    const int r = blockIdx.x*8 + wave;            // 16-row tile id (0..1023)
    const int row = r*16 + m;
    const int tg = row & (SEQ-1), b = row >> 11;
    const int c = tg >> 6, il = tg & (CHUNK-1);
    const float* hrow = Hloc + (size_t)row*STATE;
    const float* ci   = carryIn + (b*NCHUNK + c)*STATE;

    _Float16 ov[32] __attribute__((aligned(32)));
    #pragma unroll
    for (int run = 0; run < 4; ++run) {           // runs: s = run*16 + 8*half + q
        const int sbase = run*16 + 8*half;
        #pragma unroll
        for (int q = 0; q < 8; ++q) {
            const int s = sbase + q;
            const float hv = fmaf(powtab[s*(CHUNK+1) + il + 1], ci[s], hrow[s]);
            ov[run*8 + q] = (_Float16)hv;
        }
    }
    v16h* dst = (v16h*)(Hsw + ((size_t)r*32 + lane)*32);
    dst[0] = *(const v16h*)(ov);
    dst[1] = *(const v16h*)(ov + 16);
}

// ---------------------------------------------------------------------------
// Kernel 7: GEMM2  Y = H @ C^T  (K=64 -> 2 WMMA per 16-col tile), fused
// epilogue out = Y + D*x.  A operand loaded pre-swizzled (contiguous 64B).
// ---------------------------------------------------------------------------
__global__ __launch_bounds__(256) void k_gemm2(
    const _Float16* __restrict__ Hsw, const _Float16* __restrict__ Csw,
    const float* __restrict__ Dbuf, const float* __restrict__ x,
    float* __restrict__ out)
{
    const int wave = threadIdx.x >> 5, lane = threadIdx.x & 31;
    const int half = lane >> 4, m = lane & 15;
    const int r = blockIdx.x*8 + wave;            // 16-row tile id
    const v16h* hp = (const v16h*)(Hsw + ((size_t)r*32 + lane)*32);
    const v16h a0 = hp[0], a1 = hp[1];
    const v16h* cp = (const v16h*)Csw;

    for (int nt = 0; nt < 64; ++nt) {
        const v16h b0 = cp[(0*64 + nt)*32 + lane];
        const v16h b1 = cp[(1*64 + nt)*32 + lane];
        v8f acc = {};
        acc = __builtin_amdgcn_wmma_f32_16x16x32_f16(false, a0, false, b0, (short)0, acc, false, false);
        acc = __builtin_amdgcn_wmma_f32_16x16x32_f16(false, a1, false, b1, (short)0, acc, false, false);
        const int e = nt*16 + m;
        const float d = Dbuf[e];
        #pragma unroll
        for (int v = 0; v < 8; ++v) {             // D-layout: M = v + 8*half
            const size_t idx = (size_t)(r*16 + v + 8*half)*EMBED + e;
            out[idx] = fmaf(d, x[idx], acc[v]);
        }
    }
}

// ---------------------------------------------------------------------------
extern "C" void kernel_launch(void* const* d_in, const int* in_sizes, int n_in,
                              void* d_out, int out_size, void* d_ws, size_t ws_size,
                              hipStream_t stream)
{
    const float* te    = (const float*)d_in[0];
    const float* x     = (const float*)d_in[1];
    const float* Wl1   = (const float*)d_in[2];
    const float* bl1   = (const float*)d_in[3];
    const float* Wl2   = (const float*)d_in[4];
    const float* bl2   = (const float*)d_in[5];
    const float* W_A   = (const float*)d_in[6];
    const float* b_A   = (const float*)d_in[7];
    const float* W_B   = (const float*)d_in[8];
    const float* b_B   = (const float*)d_in[9];
    const float* W_C   = (const float*)d_in[10];
    const float* b_C   = (const float*)d_in[11];
    const float* W_D   = (const float*)d_in[12];
    const float* b_D   = (const float*)d_in[13];
    const float* W_dl  = (const float*)d_in[14];
    const float* b_dl  = (const float*)d_in[15];
    float* out = (float*)d_out;

    // carve workspace (all offsets 256B aligned)
    char* ws = (char*)d_ws;
    size_t off = 0;
    auto carve = [&](size_t bytes) { void* p = ws + off; off = (off + bytes + 255) & ~(size_t)255; return p; };
    float*    tbuf     = (float*)   carve(T_DIM*4);
    float*    deltaBuf = (float*)   carve(STATE*4);
    float*    abarBuf  = (float*)   carve(STATE*4);
    float*    Dbuf     = (float*)   carve(EMBED*4);
    float*    powtab   = (float*)   carve(STATE*(CHUNK+1)*4);
    _Float16* Bsw      = (_Float16*)carve((size_t)EMBED*STATE*2);          // 128 KB
    _Float16* Csw      = (_Float16*)carve((size_t)EMBED*STATE*2);          // 128 KB
    float*    CE       = (float*)   carve((size_t)BATCH*NCHUNK*STATE*4);
    float*    carryIn  = (float*)   carve((size_t)BATCH*NCHUNK*STATE*4);
    _Float16* Hsw      = (_Float16*)carve((size_t)NROWS*STATE*2);          // 2 MB
    float*    U        = (float*)   carve((size_t)NROWS*STATE*4);          // 4 MB
    float*    Hloc     = (float*)   carve((size_t)NROWS*STATE*4);          // 4 MB
    (void)ws_size; (void)in_sizes; (void)n_in; (void)out_size;

    k_hyper <<<1,   256, 0, stream>>>(te, Wl1, bl1, Wl2, bl2, W_A, b_A,
                                      W_dl, b_dl, W_D, b_D,
                                      tbuf, deltaBuf, abarBuf, powtab, Dbuf);
    k_BC    <<<512, 256, 0, stream>>>(tbuf, deltaBuf, W_B, b_B, W_C, b_C, Bsw, Csw);
    k_gemm1 <<<128, 256, 0, stream>>>(x, Bsw, U);
    k_scan1 <<<64,  256, 0, stream>>>(U, abarBuf, Hloc, CE);
    k_scan2 <<<1,   512, 0, stream>>>(CE, powtab, carryIn);
    k_fixswz<<<128, 256, 0, stream>>>(Hloc, powtab, carryIn, Hsw);
    k_gemm2 <<<128, 256, 0, stream>>>(Hsw, Csw, Dbuf, x, out);
}
